// ConcatHeadModule_43980465111472
// MI455X (gfx1250) — matrix operand order repaired and checked by reading, not compiled, not run on hardware
//
#include <hip/hip_runtime.h>
#include <hip/hip_bf16.h>

// Problem sizes (fixed by the reference)
#define N_TOK 1024
#define L_DIM 256
#define K1    512   // 2L
#define H_DIM 512
#define H2    256

typedef __attribute__((ext_vector_type(16))) _Float16      v16h;
typedef __attribute__((ext_vector_type(8)))  float         v8f;
typedef __attribute__((ext_vector_type(4)))  unsigned int  u32x4;
typedef __attribute__((ext_vector_type(4)))  float         f32x4;

union FragAB { v16h v; u32x4 q[2]; };

// tanh via hardware TRANS ops: tanh(v) = 1 - 2/(e^{2v}+1); saturates to +/-1 on overflow.
__device__ __forceinline__ float fast_tanh(float v) {
    float t = __builtin_amdgcn_exp2f(v * 2.885390081777927f); // 2*log2(e)
    return 1.0f - 2.0f * __builtin_amdgcn_rcpf(t + 1.0f);
}

// ---------------- Stage 0: convert/transpose to f16 ----------------

// xh[i][k] = concat(lstms0, lstms1) in f16, row-major [1024][512]
__global__ void cvt_x(const float* __restrict__ l0, const float* __restrict__ l1,
                      _Float16* __restrict__ xh) {
    int idx = blockIdx.x * blockDim.x + threadIdx.x;      // 1024*512 threads
    int i = idx >> 9, k = idx & 511;
    float v = (k < L_DIM) ? l0[i * L_DIM + k] : l1[i * L_DIM + (k - L_DIM)];
    xh[idx] = (_Float16)v;
}

// WT[n][k] = W[k][n] in f16 (both W_foh and W_fom), [512][512]
__global__ void cvt_w(const float* __restrict__ Wfoh, const float* __restrict__ Wfom,
                      _Float16* __restrict__ WfohT, _Float16* __restrict__ WfomT) {
    int idx = blockIdx.x * blockDim.x + threadIdx.x;      // 512*512 threads
    int n = idx >> 9, k = idx & 511;
    WfohT[idx] = (_Float16)Wfoh[k * H_DIM + n];
    WfomT[idx] = (_Float16)Wfom[k * H_DIM + n];
}

// UtopT[n][k] = hid2[k][n], UbotT[n][k] = hid2[512+k][n]; [256][512]
__global__ void cvt_u(const float* __restrict__ hid2,
                      _Float16* __restrict__ UtopT, _Float16* __restrict__ UbotT) {
    int idx = blockIdx.x * blockDim.x + threadIdx.x;      // 256*512 threads
    int n = idx >> 9, k = idx & 511;
    UtopT[idx] = (_Float16)hid2[k * H2 + n];
    UbotT[idx] = (_Float16)hid2[(H_DIM + k) * H2 + n];
}

// ---------------- WMMA tile core (16x16 C tile, K-loop in steps of 32) ----------------
// A row-major f16 [*, lda] ; BT row-major f16 [N][ldb] (i.e. B column-major).
__device__ __forceinline__ v8f wmma_tile(const _Float16* __restrict__ arow,
                                         const _Float16* __restrict__ brow,
                                         int K, int half) {
    v8f acc = {};
    for (int kb = 0; kb < K; kb += 32) {
        FragAB fa, fb;
        // A 16x32 f16 layout: elems 0..7 -> K = kb + 8*half + e ; elems 8..15 -> K = kb + 16 + 8*half + e
        fa.q[0] = *(const u32x4*)(arow + kb + 8 * half);
        fa.q[1] = *(const u32x4*)(arow + kb + 16 + 8 * half);
        // B 32x16 f16 layout: elems 0..7 -> K = kb + 8*half + e ; elems 8..15 -> K = kb + 16 + 8*half + e
        fb.q[0] = *(const u32x4*)(brow + kb + 8 * half);
        fb.q[1] = *(const u32x4*)(brow + kb + 16 + 8 * half);
        acc = __builtin_amdgcn_wmma_f32_16x16x32_f16(false, fa.v, false, fb.v,
                                                     (short)0, acc, false, false);
    }
    return acc;
}

// ---------------- Stage 1: ah/am = tanh(x @ W + catBias) -> f16 [1024][512] ----------------
__global__ void gemm_xw_tanh(const _Float16* __restrict__ xh,
                             const _Float16* __restrict__ WfohT,
                             const _Float16* __restrict__ WfomT,
                             const float* __restrict__ catBias,
                             _Float16* __restrict__ ah, _Float16* __restrict__ am) {
    const int sel = blockIdx.y;                                   // 0 -> head, 1 -> mod
    const _Float16* WT = sel ? WfomT : WfohT;
    _Float16* outp = sel ? am : ah;
    const int lane = threadIdx.x & 31;
    const int lm = lane & 15, half = lane >> 4;
    const int tileId = blockIdx.x * (blockDim.x >> 5) + (threadIdx.x >> 5);
    const int tileN = tileId & 31;           // 512/16 = 32 column tiles
    const int tileM = tileId >> 5;           // 64 row tiles
    const int row0 = tileM * 16, col0 = tileN * 16;

    v8f acc = wmma_tile(xh + (size_t)(row0 + lm) * K1,
                        WT + (size_t)(col0 + lm) * K1, K1, half);

    const int col = col0 + lm;
    const float bias = catBias[sel * H_DIM + col];
    #pragma unroll
    for (int r = 0; r < 8; ++r) {
        int row = row0 + half * 8 + r;                    // C layout: VGPR r -> M = 8*half + r
        outp[(size_t)row * H_DIM + col] = (_Float16)fast_tanh(acc[r] + bias);
    }
}

// ---------------- Stage 2: A = ah@Utop, B = am@Ubot -> f32 [1024][256] ----------------
__global__ void gemm_hu(const _Float16* __restrict__ ah, const _Float16* __restrict__ am,
                        const _Float16* __restrict__ UtopT, const _Float16* __restrict__ UbotT,
                        float* __restrict__ Aacc, float* __restrict__ Bacc) {
    const int sel = blockIdx.y;
    const _Float16* Ain = sel ? am : ah;
    const _Float16* BT  = sel ? UbotT : UtopT;
    float* outp = sel ? Bacc : Aacc;
    const int lane = threadIdx.x & 31;
    const int lm = lane & 15, half = lane >> 4;
    const int tileId = blockIdx.x * (blockDim.x >> 5) + (threadIdx.x >> 5);
    const int tileN = tileId & 15;           // 256/16 = 16 column tiles
    const int tileM = tileId >> 4;           // 64 row tiles
    const int row0 = tileM * 16, col0 = tileN * 16;

    v8f acc = wmma_tile(Ain + (size_t)(row0 + lm) * H_DIM,
                        BT + (size_t)(col0 + lm) * H_DIM, H_DIM, half);

    const int col = col0 + lm;
    #pragma unroll
    for (int r = 0; r < 8; ++r) {
        int row = row0 + half * 8 + r;
        outp[(size_t)row * H2 + col] = acc[r];
    }
}

// ---------------- Stage 3: pairwise tanh reduction (TRANS-bound) ----------------
// out[i][j] = sum_h tanh(A[i][h] + B[j][h] + bias[h]) * w[h] + c
// 32x32 output tile per 256-thread block; 2x2 register blocking per thread.
__global__ void pairwise(const float* __restrict__ A, const float* __restrict__ B,
                         const float* __restrict__ hid2Bias, const float* __restrict__ wvec,
                         const float* __restrict__ outBias, float* __restrict__ out) {
    __shared__ float As[32][132];   // 132 % 64 == 4 -> conflict-free strided access
    __shared__ float Bs[32][132];
    __shared__ float Ws[H2];
    __shared__ float biasS[H2];

    const int tid = threadIdx.x;
    const int iBase = blockIdx.y * 32;
    const int jBase = blockIdx.x * 32;
    const float* Ag = A + (size_t)iBase * H2;
    const float* Bg = B + (size_t)jBase * H2;

    if (tid < H2) { Ws[tid] = wvec[tid]; biasS[tid] = hid2Bias[tid]; }

    const int tx = tid & 15, ty = tid >> 4;
    float acc00 = 0.f, acc01 = 0.f, acc10 = 0.f, acc11 = 0.f;

    for (int hb = 0; hb < H2; hb += 128) {
        __syncthreads();
        // stage 32x128 chunks of A (with bias pre-added) and B into LDS, 128-bit loads
        for (int idx = tid; idx < 32 * 32; idx += 256) {
            int r = idx >> 5, c4 = (idx & 31) << 2;
            int h = hb + c4;
            f32x4 av = *(const f32x4*)(Ag + r * H2 + h);
            av.x += biasS[h];     av.y += biasS[h + 1];
            av.z += biasS[h + 2]; av.w += biasS[h + 3];
            *(f32x4*)&As[r][c4] = av;
            *(f32x4*)&Bs[r][c4] = *(const f32x4*)(Bg + r * H2 + h);
        }
        __syncthreads();

        const float* Ar0 = As[ty];
        const float* Ar1 = As[ty + 16];
        const float* Br0 = Bs[tx];
        const float* Br1 = Bs[tx + 16];
        #pragma unroll 4
        for (int h = 0; h < 128; ++h) {
            float wv = Ws[hb + h];
            float a0 = Ar0[h], a1 = Ar1[h];
            float b0 = Br0[h], b1 = Br1[h];
            acc00 = fmaf(fast_tanh(a0 + b0), wv, acc00);
            acc01 = fmaf(fast_tanh(a0 + b1), wv, acc01);
            acc10 = fmaf(fast_tanh(a1 + b0), wv, acc10);
            acc11 = fmaf(fast_tanh(a1 + b1), wv, acc11);
        }
    }

    const float ob = outBias[0];
    out[(size_t)(iBase + ty)      * N_TOK + jBase + tx]      = acc00 + ob;
    out[(size_t)(iBase + ty)      * N_TOK + jBase + tx + 16] = acc01 + ob;
    out[(size_t)(iBase + ty + 16) * N_TOK + jBase + tx]      = acc10 + ob;
    out[(size_t)(iBase + ty + 16) * N_TOK + jBase + tx + 16] = acc11 + ob;
}

extern "C" void kernel_launch(void* const* d_in, const int* in_sizes, int n_in,
                              void* d_out, int out_size, void* d_ws, size_t ws_size,
                              hipStream_t stream) {
    (void)in_sizes; (void)n_in; (void)out_size; (void)ws_size;
    const float* lstms0   = (const float*)d_in[0];
    const float* lstms1   = (const float*)d_in[1];
    const float* W_foh    = (const float*)d_in[2];
    const float* W_fom    = (const float*)d_in[3];
    const float* catBias  = (const float*)d_in[4];
    const float* hid2     = (const float*)d_in[5];
    const float* hid2Bias = (const float*)d_in[6];
    const float* outLayer = (const float*)d_in[7];
    const float* outBias  = (const float*)d_in[8];
    float* out = (float*)d_out;

    // workspace layout (all sizes multiples of 256B)
    char* w = (char*)d_ws;
    _Float16* xh     = (_Float16*)(w);                                      // 1024*512 f16
    _Float16* WfohT  = (_Float16*)(w + 1048576);                            // 512*512 f16
    _Float16* WfomT  = (_Float16*)(w + 1048576 + 524288);
    _Float16* UtopT  = (_Float16*)(w + 1048576 + 2 * 524288);               // 256*512 f16
    _Float16* UbotT  = (_Float16*)(w + 1048576 + 2 * 524288 + 262144);
    _Float16* ah     = (_Float16*)(w + 2621440);                            // 1024*512 f16
    _Float16* am     = (_Float16*)(w + 2621440 + 1048576);
    float*    Aacc   = (float*)   (w + 4718592);                            // 1024*256 f32
    float*    Bacc   = (float*)   (w + 4718592 + 1048576);

    // Stage 0: conversions
    cvt_x<<<dim3((N_TOK * K1) / 256), dim3(256), 0, stream>>>(lstms0, lstms1, xh);
    cvt_w<<<dim3((K1 * H_DIM) / 256), dim3(256), 0, stream>>>(W_foh, W_fom, WfohT, WfomT);
    cvt_u<<<dim3((H2 * K1) / 256),    dim3(256), 0, stream>>>(hid2, UtopT, UbotT);

    // Stage 1: ah/am  (64x32 tiles, 4 waves/block -> 512 blocks, y = {head,mod})
    gemm_xw_tanh<<<dim3(512, 2), dim3(128), 0, stream>>>(xh, WfohT, WfomT, catBias, ah, am);

    // Stage 2: A/B    (64x16 tiles, 4 waves/block -> 256 blocks)
    gemm_hu<<<dim3(256, 2), dim3(128), 0, stream>>>(ah, am, UtopT, UbotT, Aacc, Bacc);

    // Stage 3: pairwise tanh reduction over 1024x1024 outputs
    pairwise<<<dim3(N_TOK / 32, N_TOK / 32), dim3(256), 0, stream>>>(
        Aacc, Bacc, hid2Bias, outLayer, outBias, out);
}
